// model_55654186222336
// MI455X (gfx1250) — matrix operand order repaired and checked by reading, not compiled, not run on hardware
//
#include <hip/hip_runtime.h>
#include <hip/hip_bf16.h>

// ---------------------------------------------------------------------------
// Problem constants (from reference): B=4, N=2048, DIM=1024, HID=1024
// ---------------------------------------------------------------------------
#define B_   4
#define N_   2048
#define DIM_ 1024
#define HID_ 1024
#define H3_  3072           // 3*HID
#define M_   (B_ * N_)      // 8192 rows for QKV GEMM
#define BN_  32             // k-tile rows per flash iteration

typedef __attribute__((ext_vector_type(16))) __bf16 v16bf;
typedef __attribute__((ext_vector_type(8)))  __bf16 bf16x8;   // 16 bytes
typedef __attribute__((ext_vector_type(2)))  __bf16 bf16x2;   // 4 bytes
typedef __attribute__((ext_vector_type(8)))  float  v8f;
typedef __attribute__((ext_vector_type(4)))  unsigned int u32x4;
typedef __attribute__((ext_vector_type(8)))  int    i32x8;
typedef __attribute__((ext_vector_type(4)))  int    i32x4;

static __device__ inline v8f vzero8() {
    v8f z = {0.f, 0.f, 0.f, 0.f, 0.f, 0.f, 0.f, 0.f};
    return z;
}

// ---------------------------------------------------------------------------
// Tensor Data Mover: 2D tile load Global -> LDS.
// D# layout per CDNA5 ISA ch.8: group0 = {count/type/addr}, group1 = dims.
// data_size code 1 == 2-byte elements (bf16). Dims in element units.
// Toolchain here is the 6-arg clang-23 flavor:
//   (uint32x4 g0, int32x8 g1, int32x4 g2, int32x4 g3, int32x8 g4, i32 cpol)
// ---------------------------------------------------------------------------
static __device__ inline void tdm_load_2d_bf16(unsigned lds_off, const void* gptr,
                                               unsigned tensor_d0, unsigned tensor_d1,
                                               unsigned tile_d0, unsigned tile_d1,
                                               unsigned stride0) {
    unsigned long long ga = (unsigned long long)(uintptr_t)gptr;
    u32x4 g0;
    g0[0] = 1u;                                   // count=1, user descriptor
    g0[1] = lds_off;                              // lds_addr (bytes)
    g0[2] = (unsigned)(ga & 0xffffffffu);         // global_addr[31:0]
    g0[3] = (unsigned)((ga >> 32) & 0x1ffffffu)   // global_addr[56:32]
          | (2u << 30);                           // type=2 ("image")
    i32x8 g1;
    g1[0] = (int)(1u << 16);                      // workgroup_mask=0, data_size=1 (2B)
    g1[1] = (int)((tensor_d0 & 0xffffu) << 16);   // abar=0 | tensor_dim0[15:0]
    g1[2] = (int)((tensor_d0 >> 16) & 0xffffu)    // tensor_dim0[31:16]
          | (int)((tensor_d1 & 0xffffu) << 16);   // tensor_dim1[15:0]
    g1[3] = (int)((tensor_d1 >> 16) & 0xffffu)    // tensor_dim1[31:16]
          | (int)((tile_d0 & 0xffffu) << 16);     // tile_dim0
    g1[4] = (int)(tile_d1 & 0xffffu);             // tile_dim1 | tile_dim2=0
    g1[5] = (int)stride0;                         // tensor_dim0_stride[31:0]
    g1[6] = 0;                                    // stride0[47:32] | dim1_stride lo
    g1[7] = 0;
    i32x4 z4 = {0, 0, 0, 0};
    i32x8 z8 = {0, 0, 0, 0, 0, 0, 0, 0};
    __builtin_amdgcn_tensor_load_to_lds(g0, g1, z4, z4, z8, 0);
}

static __device__ inline unsigned lds_offset_of(const void* p) {
    return (unsigned)(uintptr_t)p;   // generic LDS addr: low 32 bits = LDS offset
}

// ---------------------------------------------------------------------------
// WMMA fragment loaders (wave32, 16x16x32 bf16 shapes per ISA 7.12.2)
// A (16Mx32K): lane l -> m = l%16 ; K-half base = (l<16 ? 0 : 16)
// ---------------------------------------------------------------------------
static __device__ inline v16bf load_a_frag(const __bf16* src, int ld) {
    int l = threadIdx.x & 31;
    int m = l & 15;
    int kb = (l < 16) ? 0 : 16;
    v16bf a;
#pragma unroll
    for (int i = 0; i < 8; ++i) {
        int k = kb + 2 * i;
        bf16x2 p = *(const bf16x2*)(src + (size_t)m * ld + k);
        a[2 * i]     = p[0];
        a[2 * i + 1] = p[1];
    }
    return a;
}

// B (32Kx16N), source stored K-contiguous per column: element (k,n) at srcT[n*ld+k]
static __device__ inline v16bf load_b_frag_t(const __bf16* srcT, int ld) {
    int l = threadIdx.x & 31;
    int n = l & 15;
    int kb = (l < 16) ? 0 : 16;
    v16bf b;
#pragma unroll
    for (int v = 0; v < 8; ++v) {
        bf16x2 p = *(const bf16x2*)(srcT + (size_t)n * ld + kb + 2 * v);
        b[2 * v]     = p[0];
        b[2 * v + 1] = p[1];
    }
    return b;
}

// B (32Kx16N), source stored row-major: element (k,n) at src[k*ld + n]
static __device__ inline v16bf load_b_frag(const __bf16* src, int ld) {
    int l = threadIdx.x & 31;
    int n = l & 15;
    int kb = (l < 16) ? 0 : 16;
    v16bf b;
#pragma unroll
    for (int v = 0; v < 8; ++v) {
        b[2 * v]     = src[(size_t)(kb + 2 * v)     * ld + n];
        b[2 * v + 1] = src[(size_t)(kb + 2 * v + 1) * ld + n];
    }
    return b;
}

// ---------------------------------------------------------------------------
// Kernel 1: fp32 -> bf16 convert (grid-stride)
// ---------------------------------------------------------------------------
__global__ void f32_to_bf16_kernel(const float* __restrict__ src,
                                   __bf16* __restrict__ dst, int n) {
    int i = blockIdx.x * blockDim.x + threadIdx.x;
    int stride = gridDim.x * blockDim.x;
    for (; i < n; i += stride) dst[i] = (__bf16)src[i];
}

// ---------------------------------------------------------------------------
// Kernel 2: QKV GEMM  qkv[M,3H] = xb[M,DIM] @ Wb[DIM,3H] + bias
// 256 threads (8 waves), tile 64(M) x 128(N), K-step 32; wave = 16x64 slab.
// ---------------------------------------------------------------------------
__global__ __launch_bounds__(256) void qkv_gemm_kernel(
    const __bf16* __restrict__ xb, const __bf16* __restrict__ Wb,
    const float* __restrict__ bias,
    __bf16* __restrict__ qb, __bf16* __restrict__ kb, __bf16* __restrict__ vb) {

    __shared__ __attribute__((aligned(16))) __bf16 Als[64 * 32];   // [m][k]
    __shared__ __attribute__((aligned(16))) __bf16 Bls[128 * 32];  // [n][k] transposed

    const int m0  = blockIdx.y * 64;
    const int n0  = blockIdx.x * 128;
    const int tid = threadIdx.x;
    const int wid = tid >> 5;
    const int lane = tid & 31;
    const int wm = wid & 3;
    const int wn = wid >> 2;

    v8f acc[4];
#pragma unroll
    for (int s = 0; s < 4; ++s) acc[s] = vzero8();

    for (int k0 = 0; k0 < DIM_; k0 += 32) {
        {   // A tile 64x32: one 16B chunk per thread
            int e = tid * 8;
            int r = e >> 5, c = e & 31;
            *(bf16x8*)&Als[e] =
                *(const bf16x8*)(xb + (size_t)(m0 + r) * DIM_ + k0 + c);
        }
        {   // B tile 32x128 -> transposed [n][k]
            int kk = tid >> 3;
            int nb = (tid & 7) * 16;
            const __bf16* src = Wb + (size_t)(k0 + kk) * H3_ + n0 + nb;
#pragma unroll
            for (int j = 0; j < 16; ++j)
                Bls[(nb + j) * 32 + kk] = src[j];
        }
        __syncthreads();

        v16bf a = load_a_frag(&Als[wm * 16 * 32], 32);
#pragma unroll
        for (int s = 0; s < 4; ++s) {
            v16bf bb = load_b_frag_t(&Bls[(wn * 64 + s * 16) * 32], 32);
            acc[s] = __builtin_amdgcn_wmma_f32_16x16x32_bf16(
                false, a, false, bb, (short)0, acc[s], false, false);
        }
        __syncthreads();
    }

    const int rlo   = lane & 15;
    const int hiofs = (lane >> 4) * 8;
#pragma unroll
    for (int s = 0; s < 4; ++s) {
#pragma unroll
        for (int v = 0; v < 8; ++v) {
            int gm = m0 + wm * 16 + v + hiofs;
            int gc = n0 + wn * 64 + s * 16 + rlo;
            float val = acc[s][v] + bias[gc];
            int reg = gc >> 10;
            int c   = gc & 1023;
            __bf16* dst = (reg == 0) ? qb : ((reg == 1) ? kb : vb);
            dst[(size_t)gm * HID_ + c] = (__bf16)val;
        }
    }
}

// ---------------------------------------------------------------------------
// Kernel 3: flash attention with TDM-staged, double-buffered K/V tiles.
// Grid: (N/16, B). Block: 256 threads = 8 waves.
// Workgroup owns 16 query rows; wave w owns output h-slice [w*128, w*128+128).
// Per 32-row k-tile: partial S via WMMA, LDS reduce, online softmax, O += P@V.
// ---------------------------------------------------------------------------
__global__ __launch_bounds__(256) void attn_flash_kernel(
    const __bf16* __restrict__ qb, const __bf16* __restrict__ kbm,
    const __bf16* __restrict__ vbm, float* __restrict__ out) {

    __shared__ __attribute__((aligned(16))) __bf16 Kls[2][BN_ * HID_]; // 2x64KB
    __shared__ __attribute__((aligned(16))) __bf16 Vls[2][BN_ * HID_]; // 2x64KB
    __shared__ __attribute__((aligned(16))) float  Sp[8][16][BN_];     // 16KB
    __shared__ __attribute__((aligned(16))) __bf16 P[16][BN_];         // 1KB
    __shared__ float mrun[16], lrun[16], alphas[16];

    const int bat = blockIdx.y;
    const int q0  = blockIdx.x * 16;
    const int tid = threadIdx.x;
    const int wid = tid >> 5;
    const int lane = tid & 31;
    const int h0  = wid * 128;

    const __bf16* Q = qb  + (size_t)(bat * N_ + q0) * HID_;
    const __bf16* K = kbm + (size_t)bat * N_ * HID_;
    const __bf16* V = vbm + (size_t)bat * N_ * HID_;

    if (tid < 16) { mrun[tid] = -3.0e38f; lrun[tid] = 0.0f; }

    // Hoist Q fragments (this wave's 128-wide h slice) into registers.
    v16bf qa[4];
#pragma unroll
    for (int kc = 0; kc < 4; ++kc)
        qa[kc] = load_a_frag(Q + h0 + kc * 32, HID_);

    v8f acc[8];
#pragma unroll
    for (int s = 0; s < 8; ++s) acc[s] = vzero8();

    const float scale = 0.03125f;  // 1/sqrt(1024)
    const int rlo   = lane & 15;
    const int hiofs = (lane >> 4) * 8;
    const int T = N_ / BN_;        // 64 k-tiles

    // Preload tile 0 via TDM (wave 0 issues; TENSORcnt tracks completion).
    if (wid == 0) {
        tdm_load_2d_bf16(lds_offset_of(&Kls[0][0]), K,
                         HID_, N_, HID_, BN_, HID_);
        tdm_load_2d_bf16(lds_offset_of(&Vls[0][0]), V,
                         HID_, N_, HID_, BN_, HID_);
    }
    __syncthreads();

    for (int t = 0; t < T; ++t) {
        const int k0 = t * BN_;
        if (wid == 0) __builtin_amdgcn_s_wait_tensorcnt(0);
        __syncthreads();   // buf[t&1] ready for everyone; prev reads retired

        const __bf16* Kb = &Kls[t & 1][0];
        const __bf16* Vb = &Vls[t & 1][0];

        // Kick off DMA for next tile into the other buffer (overlaps compute).
        if (wid == 0 && (t + 1) < T) {
            tdm_load_2d_bf16(lds_offset_of(&Kls[(t + 1) & 1][0]),
                             K + (size_t)(k0 + BN_) * HID_,
                             HID_, N_, HID_, BN_, HID_);
            tdm_load_2d_bf16(lds_offset_of(&Vls[(t + 1) & 1][0]),
                             V + (size_t)(k0 + BN_) * HID_,
                             HID_, N_, HID_, BN_, HID_);
        }

        // ---- 1. partial S over this wave's 128-wide h slice -------------
        v8f sa[2];
#pragma unroll
        for (int ct = 0; ct < 2; ++ct) sa[ct] = vzero8();
#pragma unroll
        for (int kc = 0; kc < 4; ++kc) {
#pragma unroll
            for (int ct = 0; ct < 2; ++ct) {
                // B = K^T: element (h,kcol) at Kb[(ct*16+kcol)*HID + h0+kc*32+h]
                v16bf bb = load_b_frag_t(
                    Kb + (size_t)(ct * 16) * HID_ + h0 + kc * 32, HID_);
                sa[ct] = __builtin_amdgcn_wmma_f32_16x16x32_bf16(
                    false, qa[kc], false, bb, (short)0, sa[ct], false, false);
            }
        }
#pragma unroll
        for (int ct = 0; ct < 2; ++ct)
#pragma unroll
            for (int v = 0; v < 8; ++v)
                Sp[wid][v + hiofs][ct * 16 + rlo] = sa[ct][v];
        __syncthreads();

        // ---- 2. reduce 8 partials -> Sp[0] (512 elems / 256 threads) ----
#pragma unroll
        for (int j = 0; j < 2; ++j) {
            int e = tid * 2 + j;
            int r = e >> 5, c = e & (BN_ - 1);
            float s = 0.f;
#pragma unroll
            for (int w = 0; w < 8; ++w) s += Sp[w][r][c];
            Sp[0][r][c] = s;
        }
        __syncthreads();

        // ---- 3. online softmax row update (16 threads, one row each) ----
        if (tid < 16) {
            int r = tid;
            float mold = mrun[r], mnew = mold;
            for (int c = 0; c < BN_; ++c)
                mnew = fmaxf(mnew, Sp[0][r][c] * scale);
            float al = __expf(mold - mnew);
            float sum = 0.f;
            for (int c = 0; c < BN_; ++c) {
                float p = __expf(Sp[0][r][c] * scale - mnew);
                P[r][c] = (__bf16)p;
                sum += p;
            }
            lrun[r]   = al * lrun[r] + sum;
            mrun[r]   = mnew;
            alphas[r] = al;
        }
        __syncthreads();

        // ---- 4. rescale running O accumulators --------------------------
        {
            float al[8];
#pragma unroll
            for (int v = 0; v < 8; ++v) al[v] = alphas[v + hiofs];
#pragma unroll
            for (int s = 0; s < 8; ++s)
#pragma unroll
                for (int v = 0; v < 8; ++v) acc[s][v] *= al[v];
        }

        // ---- 5. O += P @ V over this wave's h slice (K-dim = 32) --------
        {
            v16bf a = load_a_frag(&P[0][0], BN_);
#pragma unroll
            for (int s = 0; s < 8; ++s) {
                v16bf bb = load_b_frag(Vb + h0 + s * 16, HID_);
                acc[s] = __builtin_amdgcn_wmma_f32_16x16x32_bf16(
                    false, a, false, bb, (short)0, acc[s], false, false);
            }
        }
        __syncthreads();   // protect P / stats / LDS buffers for next iter
    }

    // ---- finalize: divide by row sums, store fp32 output ----------------
    {
        float li[8];
#pragma unroll
        for (int v = 0; v < 8; ++v) li[v] = 1.0f / lrun[v + hiofs];
#pragma unroll
        for (int s = 0; s < 8; ++s)
#pragma unroll
            for (int v = 0; v < 8; ++v) {
                int row = v + hiofs;
                out[(size_t)(bat * N_ + q0 + row) * HID_ + h0 + s * 16 + rlo] =
                    acc[s][v] * li[v];
            }
    }
}

// ---------------------------------------------------------------------------
// Launcher.  Workspace layout (bf16), total ~70 MB:
//   xb[M*DIM] | Wb[DIM*3H] | qb[M*HID] | kb[M*HID] | vb[M*HID]
// ---------------------------------------------------------------------------
extern "C" void kernel_launch(void* const* d_in, const int* in_sizes, int n_in,
                              void* d_out, int out_size, void* d_ws, size_t ws_size,
                              hipStream_t stream) {
    const float* x    = (const float*)d_in[0];
    const float* W    = (const float*)d_in[1];
    const float* bias = (const float*)d_in[2];
    float* out = (float*)d_out;

    __bf16* xb = (__bf16*)d_ws;
    __bf16* Wb = xb + (size_t)M_ * DIM_;
    __bf16* qb = Wb + (size_t)DIM_ * H3_;
    __bf16* kb = qb + (size_t)M_ * HID_;
    __bf16* vb = kb + (size_t)M_ * HID_;

    f32_to_bf16_kernel<<<4096, 256, 0, stream>>>(x, xb, M_ * DIM_);
    f32_to_bf16_kernel<<<4096, 256, 0, stream>>>(W, Wb, DIM_ * H3_);

    qkv_gemm_kernel<<<dim3(H3_ / 128, M_ / 64), 256, 0, stream>>>(
        xb, Wb, bias, qb, kb, vb);

    attn_flash_kernel<<<dim3(N_ / 16, B_), 256, 0, stream>>>(qb, kb, vb, out);
}
